// Pol2Vec_37099927502994
// MI455X (gfx1250) — matrix operand order repaired and checked by reading, not compiled, not run on hardware
//
#include <hip/hip_runtime.h>
#include <hip/hip_bf16.h>

// Problem dims (fixed by the reference)
#define R_DIM 2048
#define C_DIM 2048
#define D_DIM 64
#define KB    288           // base K: 256 (z-blocks) + 16 (Gram/coeff-pairs) + 1 (const) + 15 pad
#define KT    (3 * KB)      // split-bf16: [Ah|Ah|Al] x [Bh|Bl|Bh]  -> 864
#define NIT   (KT / 32)     // 27 k-slices
#define TILE  128
#define KC    32            // K per WMMA / per LDS slice
#define LDST  40            // padded LDS row stride in bf16 elements (32 + 8)
#define NBUF  3             // prefetch distance 2

typedef __attribute__((ext_vector_type(16))) __bf16 v16bf;
typedef __attribute__((ext_vector_type(8)))  __bf16 v8bf;
typedef __attribute__((ext_vector_type(8)))  float  v8f;

// Pointer types matching the async-copy builtin's signature
typedef int v4i __attribute__((vector_size(16)));
typedef __attribute__((address_space(1))) v4i gv4i;   // global int4
typedef __attribute__((address_space(3))) v4i lv4i;   // LDS int4

union FragU { v16bf v; v8bf h[2]; };

// CDNA5 async global->LDS copy (ASYNCcnt-tracked, no VGPR staging)
#define ASYNC_CP16(gsrc, ldst)                                                  \
    __builtin_amdgcn_global_load_async_to_lds_b128(                             \
        (gv4i*)(gsrc), (lv4i*)(ldst), 0, 0)

// Async loads retire in order: waiting cnt<=4 retires the oldest slice while
// leaving the newest 4 copies in flight.
#define WAIT_ASYNC4() asm volatile("s_wait_asynccnt 0x4" ::: "memory")
#define WAIT_ASYNC0() asm volatile("s_wait_asynccnt 0x0" ::: "memory")

__device__ __forceinline__ unsigned short f2bf(float x) {
    unsigned int u = __float_as_uint(x);
    u += 0x7FFFu + ((u >> 16) & 1u);           // round-to-nearest-even
    return (unsigned short)(u >> 16);
}
__device__ __forceinline__ float bf2f(unsigned short h) {
    return __uint_as_float(((unsigned int)h) << 16);
}

// ---------------------------------------------------------------------------
// Build A3[r, 864] = [Ah | Ah | Al], base row layout:
//   k in [0,256)   : z_rows[o=k/64, r, d=k%64]
//   k in [256,272) : H[o,o',r] = sum_d z_o z_o'   (ordered pairs)
//   k == 272       : 1.0
//   k in [273,288) : 0
// ---------------------------------------------------------------------------
__global__ __launch_bounds__(64) void build_a3(const float* __restrict__ z_rows,
                                               unsigned short* __restrict__ A3) {
    const int r = blockIdx.x;
    const int d = threadIdx.x;                 // 0..63
    __shared__ float prod[16][64];

    float z[4];
#pragma unroll
    for (int o = 0; o < 4; ++o)
        z[o] = z_rows[(size_t)o * (R_DIM * D_DIM) + (size_t)r * D_DIM + d];

#pragma unroll
    for (int o = 0; o < 4; ++o)
#pragma unroll
        for (int p = 0; p < 4; ++p)
            prod[o * 4 + p][d] = z[o] * z[p];
    __syncthreads();

    unsigned short* row = A3 + (size_t)r * KT;
#pragma unroll
    for (int o = 0; o < 4; ++o) {
        float v = z[o];
        unsigned short hi = f2bf(v);
        unsigned short lo = f2bf(v - bf2f(hi));
        int k = o * 64 + d;
        row[k] = hi; row[KB + k] = hi; row[2 * KB + k] = lo;
    }
    if (d < 32) {
        float v = 0.0f;
        if (d < 16) {                          // fixed-order deterministic sum
            float s = 0.0f;
            for (int i = 0; i < 64; ++i) s += prod[d][i];
            v = s;
        } else if (d == 16) {
            v = 1.0f;
        }
        unsigned short hi = f2bf(v);
        unsigned short lo = f2bf(v - bf2f(hi));
        int k = 256 + d;
        row[k] = hi; row[KB + k] = hi; row[2 * KB + k] = lo;
    }
}

// ---------------------------------------------------------------------------
// Build B3[c, 864] = [Bh | Bl | Bh], base row layout:
//   k in [0,256)   : -2 * coeff[o] * z_cols[c, d]
//   k in [256,272) : coeff[o]*coeff[o']
//   k == 272       : ||z_cols[c]||^2
//   k in [273,288) : 0
// ---------------------------------------------------------------------------
__global__ __launch_bounds__(64) void build_b3(const float* __restrict__ col_times,
                                               const float* __restrict__ z_cols,
                                               unsigned short* __restrict__ B3) {
    const int c = blockIdx.x;
    const int d = threadIdx.x;                 // 0..63
    __shared__ float sq[64];

    const float t = col_times[c];
    float coeff[4] = {1.0f, t, t * t * 0.5f, t * t * t * (1.0f / 6.0f)};
    const float zc = z_cols[(size_t)c * D_DIM + d];
    sq[d] = zc * zc;
    __syncthreads();

    unsigned short* row = B3 + (size_t)c * KT;
#pragma unroll
    for (int o = 0; o < 4; ++o) {
        float v = -2.0f * coeff[o] * zc;
        unsigned short hi = f2bf(v);
        unsigned short lo = f2bf(v - bf2f(hi));
        int k = o * 64 + d;
        row[k] = hi; row[KB + k] = lo; row[2 * KB + k] = hi;
    }
    if (d < 32) {
        float v = 0.0f;
        if (d < 16) {
            v = coeff[d >> 2] * coeff[d & 3];
        } else if (d == 16) {                  // fixed-order deterministic sum
            float s = 0.0f;
            for (int i = 0; i < 64; ++i) s += sq[i];
            v = s;
        }
        unsigned short hi = f2bf(v);
        unsigned short lo = f2bf(v - bf2f(hi));
        int k = 256 + d;
        row[k] = hi; row[KB + k] = lo; row[2 * KB + k] = hi;
    }
}

// ---------------------------------------------------------------------------
// Main: 128x128 tile per block; dist^2 = A3 . B3^T via v_wmma_f32_16x16x32_bf16.
// Triple-buffered LDS fed by GLOBAL_LOAD_ASYNC_TO_LDS_B128 at prefetch
// distance 2, retired with thresholded s_wait_asynccnt (in-order completion).
// Fused sqrt + log-sigmoid epilogue; deterministic per-block partial sums.
// ---------------------------------------------------------------------------
__global__ __launch_bounds__(256) void dist_nll_gemm(
    const unsigned short* __restrict__ A3,
    const unsigned short* __restrict__ B3,
    const int*   __restrict__ events,
    const float* __restrict__ gamma_rows,
    const float* __restrict__ gamma_cols,
    float* __restrict__ partials) {

    __shared__ __align__(16) unsigned short lA[NBUF][TILE * LDST];
    __shared__ __align__(16) unsigned short lB[NBUF][TILE * LDST];
    __shared__ float red[256];

    const int tid  = threadIdx.x;
    const int lane = tid & 31;
    const int wave = tid >> 5;                 // 0..7
    const int wm   = wave & 3;                 // 4 row groups of 32
    const int wn   = wave >> 2;                // 2 col groups of 64
    const int rowBase = blockIdx.y * TILE;
    const int colBase = blockIdx.x * TILE;

    const int g  = lane >> 4;                  // lane half
    const int li = lane & 15;

    // This thread's two (row, k-offset) staging slots within a 128x32 slice
    const int rr0 = tid >> 2;                  // rows 0..63
    const int kk0 = (tid & 3) * 8;
    const int rr1 = (tid + 256) >> 2;          // rows 64..127

    const unsigned short* gA0 = A3 + (size_t)(rowBase + rr0) * KT + kk0;
    const unsigned short* gA1 = A3 + (size_t)(rowBase + rr1) * KT + kk0;
    const unsigned short* gB0 = B3 + (size_t)(colBase + rr0) * KT + kk0;
    const unsigned short* gB1 = B3 + (size_t)(colBase + rr1) * KT + kk0;
    const int sl0 = rr0 * LDST + kk0;
    const int sl1 = rr1 * LDST + kk0;

#define ISSUE_SLICE(buf, slice)                                                 \
    do {                                                                        \
        const size_t kb_ = (size_t)(slice) * KC;                                \
        ASYNC_CP16(gA0 + kb_, &lA[(buf)][sl0]);                                 \
        ASYNC_CP16(gB0 + kb_, &lB[(buf)][sl0]);                                 \
        ASYNC_CP16(gA1 + kb_, &lA[(buf)][sl1]);                                 \
        ASYNC_CP16(gB1 + kb_, &lB[(buf)][sl1]);                                 \
    } while (0)

    v8f acc[2][4];
#pragma unroll
    for (int mt = 0; mt < 2; ++mt)
#pragma unroll
        for (int nt = 0; nt < 4; ++nt)
#pragma unroll
            for (int e = 0; e < 8; ++e) acc[mt][nt][e] = 0.0f;

    // Prologue: prefetch slices 0 and 1 (8 async ops outstanding)
    ISSUE_SLICE(0, 0);
    ISSUE_SLICE(1, 1);

    for (int it = 0; it < NIT; ++it) {
        const int cur = it % NBUF;

        // In-order retirement: cnt<=4 guarantees slice `it` landed while
        // slice it+1's copies stay in flight. Last slice: drain fully.
        if (it + 1 < NIT) WAIT_ASYNC4(); else WAIT_ASYNC0();
        __syncthreads();                       // all waves' slice-it writes visible

        // Fragment loads from buf[cur]
        // A (16x32 bf16): lane half g holds K chunks {g*8..g*8+7, 16+g*8..}
        FragU afrag[2];
#pragma unroll
        for (int mt = 0; mt < 2; ++mt) {
            int m = wm * 32 + mt * 16 + li;
            afrag[mt].h[0] = *(const v8bf*)(&lA[cur][m * LDST + g * 8]);
            afrag[mt].h[1] = *(const v8bf*)(&lA[cur][m * LDST + 16 + g * 8]);
        }
        // B (32x16 bf16): lane half g holds K chunks {g*8..g*8+7, 16+g*8..}
        FragU bfrag[4];
#pragma unroll
        for (int nt = 0; nt < 4; ++nt) {
            int n = wn * 64 + nt * 16 + li;
            bfrag[nt].h[0] = *(const v8bf*)(&lB[cur][n * LDST + g * 8]);
            bfrag[nt].h[1] = *(const v8bf*)(&lB[cur][n * LDST + 16 + g * 8]);
        }

        // Prefetch slice it+2 into the buffer consumed in the previous iteration;
        // its readers finished before this iteration's barrier.
        if (it + 2 < NIT) ISSUE_SLICE((it + 2) % NBUF, it + 2);

#pragma unroll
        for (int mt = 0; mt < 2; ++mt)
#pragma unroll
            for (int nt = 0; nt < 4; ++nt)
                acc[mt][nt] = __builtin_amdgcn_wmma_f32_16x16x32_bf16(
                    false, afrag[mt].v, false, bfrag[nt].v,
                    (short)0, acc[mt][nt], false, false);
    }
#undef ISSUE_SLICE

    // Epilogue: C/D layout -> (m = e + 8*g, n = li); nll = -log_sigmoid(sign*logit)
    float nll = 0.0f;
#pragma unroll
    for (int mt = 0; mt < 2; ++mt) {
#pragma unroll
        for (int nt = 0; nt < 4; ++nt) {
            const int r0 = rowBase + wm * 32 + mt * 16 + g * 8;
            const int c  = colBase + wn * 64 + nt * 16 + li;
            const float gc = gamma_cols[c];
#pragma unroll
            for (int e = 0; e < 8; ++e) {
                const int r = r0 + e;
                float d2    = acc[mt][nt][e];
                float dist  = sqrtf(fmaxf(d2, 0.0f));
                float logit = gamma_rows[r] + gc - dist;
                float x     = (events[(size_t)r * C_DIM + c] == 1) ? logit : -logit;
                // -log_sigmoid(x) = log1p(exp(-|x|)) - min(x, 0)
                nll += log1pf(__expf(-fabsf(x))) - fminf(x, 0.0f);
            }
        }
    }

    red[tid] = nll;
    __syncthreads();
    for (int s = 128; s > 0; s >>= 1) {        // fixed-order tree: deterministic
        if (tid < s) red[tid] += red[tid + s];
        __syncthreads();
    }
    if (tid == 0) partials[blockIdx.y * gridDim.x + blockIdx.x] = red[0];
}

__global__ __launch_bounds__(256) void final_reduce(const float* __restrict__ partials,
                                                    float* __restrict__ out) {
    __shared__ float red[256];
    const int t = threadIdx.x;
    red[t] = partials[t];
    __syncthreads();
    for (int s = 128; s > 0; s >>= 1) {
        if (t < s) red[t] += red[t + s];
        __syncthreads();
    }
    if (t == 0) out[0] = red[0];
}

// ---------------------------------------------------------------------------
extern "C" void kernel_launch(void* const* d_in, const int* in_sizes, int n_in,
                              void* d_out, int out_size, void* d_ws, size_t ws_size,
                              hipStream_t stream) {
    (void)in_sizes; (void)n_in; (void)out_size; (void)ws_size;

    const int*   events     = (const int*)  d_in[0];
    const float* col_times  = (const float*)d_in[1];
    const float* z_rows     = (const float*)d_in[2];
    const float* z_cols     = (const float*)d_in[3];
    const float* gamma_rows = (const float*)d_in[4];
    const float* gamma_cols = (const float*)d_in[5];

    const size_t panelBytes = (size_t)R_DIM * KT * sizeof(unsigned short); // 3,538,944
    unsigned short* A3 = (unsigned short*)d_ws;
    unsigned short* B3 = (unsigned short*)((char*)d_ws + panelBytes);
    float* partials    = (float*)((char*)d_ws + 2 * panelBytes);

    build_a3<<<R_DIM, 64, 0, stream>>>(z_rows, A3);
    build_b3<<<C_DIM, 64, 0, stream>>>(col_times, z_cols, B3);

    dim3 grid(C_DIM / TILE, R_DIM / TILE);     // 16 x 16 = 256 blocks
    dist_nll_gemm<<<grid, 256, 0, stream>>>(A3, B3, events, gamma_rows, gamma_cols,
                                            partials);
    final_reduce<<<1, 256, 0, stream>>>(partials, (float*)d_out);
}
